// Multi_Self_Attention_36885179138226
// MI455X (gfx1250) — compile-verified, matrix operand order checked by reading, not run on hardware
//
#include <hip/hip_runtime.h>
#include <hip/hip_bf16.h>
#include <math.h>

// Problem constants (reference: B=2, S=2048, E=1024, H=16, HD=64)
#define BDIM   2
#define SDIM   2048
#define EDIM   1024
#define HDIM   16
#define HDDIM  64
#define MROWS  (BDIM * SDIM)   // 4096 rows for the [BS, E] GEMMs

typedef __bf16 bf16;
typedef __attribute__((ext_vector_type(16))) __bf16 v16bf;
typedef __attribute__((ext_vector_type(8)))  __bf16 v8bf;
typedef __attribute__((ext_vector_type(8)))  float  v8f;
typedef __attribute__((ext_vector_type(4)))  int    v4i;

// Address-space-qualified int4 for the async global->LDS builtin, whose
// signature (from the compiler diagnostic) is:
//   void (v4i __as1*, v4i __as3*, constant int offset, constant int cpol)
typedef __attribute__((address_space(1))) v4i as1_v4i;
typedef __attribute__((address_space(3))) v4i as3_v4i;

#if __has_builtin(__builtin_amdgcn_global_load_async_to_lds_b128) && \
    __has_builtin(__builtin_amdgcn_s_wait_asynccnt)
#define HAVE_ASYNC_LDS 1
#else
#define HAVE_ASYNC_LDS 0
#endif

// Copy 16 bytes global -> LDS. Async (ASYNCcnt-tracked, no VGPR round trip)
// when the gfx1250 builtin is available, plain load+store otherwise.
__device__ __forceinline__ void g2l_b128(bf16* lds, const bf16* g) {
#if HAVE_ASYNC_LDS
  // generic->as1: global generic addresses equal as1 addresses.
  // generic->as3: LDS offset is the low 32 bits of the generic address
  // (inverse of the compiler's as3->generic addrspacecast).
  __builtin_amdgcn_global_load_async_to_lds_b128(
      (as1_v4i*)(unsigned long long)(size_t)g,
      (as3_v4i*)(unsigned int)(size_t)lds, 0, 0);
#else
  *(v8bf*)lds = *(const v8bf*)g;
#endif
}
__device__ __forceinline__ void g2l_fence() {
#if HAVE_ASYNC_LDS
  __builtin_amdgcn_s_wait_asynccnt(0);
#endif
}

__device__ __forceinline__ v16bf combine16(v8bf lo, v8bf hi) {
  return __builtin_shufflevector(lo, hi, 0,1,2,3,4,5,6,7,8,9,10,11,12,13,14,15);
}

__device__ __forceinline__ v8f wmma_bf16(v16bf a, v16bf b, v8f c) {
  // D = A(16x32 bf16) * B(32x16 bf16) + C(16x16 f32)
  return __builtin_amdgcn_wmma_f32_16x16x32_bf16(false, a, false, b, (short)0, c,
                                                 false, false);
}

// ---------------------------------------------------------------------------
// f32 -> bf16 elementwise convert
// ---------------------------------------------------------------------------
__global__ void cvt_bf16_kernel(const float* __restrict__ src,
                                bf16* __restrict__ dst, int n) {
  int i = blockIdx.x * blockDim.x + threadIdx.x;
  int stride = gridDim.x * blockDim.x;
  for (; i < n; i += stride) dst[i] = (bf16)src[i];
}

// ---------------------------------------------------------------------------
// sumv[bh][d] = sum_k V[bh][k][d]  (for the reference's softmax + 1e-8 term)
// ---------------------------------------------------------------------------
__global__ void sumv_kernel(const bf16* __restrict__ v, float* __restrict__ sumv) {
  const int bh = blockIdx.x;       // 0..31
  const int d  = threadIdx.x;      // 0..63
  const bf16* p = v + (size_t)bh * SDIM * HDDIM + d;
  float s = 0.f;
  for (int k = 0; k < SDIM; ++k) s += (float)p[(size_t)k * HDDIM];
  sumv[bh * HDDIM + d] = s;
}

// ---------------------------------------------------------------------------
// Tiled bf16 WMMA GEMM: C[M,N] = (A[M,K] * Bw[N,K]^T + bias[N]) * alpha
// Block tile 128x128, BK=32, 128 threads = 4 waves arranged 2(M) x 2(N);
// each wave owns a 64x64 tile = 4x4 grid of 16x16 WMMAs (16 accumulators),
// giving a 1:1 ds_load_b128 : v_wmma ratio in the inner loop.
// MODE 0: write f32 row-major [M,N]      (final projection -> d_out)
// MODE 1: write bf16 scattered [B,H,S,HD] (QKV projections; alpha folds the
//         1/sqrt(HD) attention scale into Q)
// ---------------------------------------------------------------------------
template <int MODE>
__global__ __launch_bounds__(128)
void gemm_bf16_kernel(const bf16* __restrict__ A, const bf16* __restrict__ Bw,
                      const float* __restrict__ bias, float alpha,
                      void* __restrict__ outv, int M, int N, int K) {
  __shared__ __align__(16) bf16 As[128][40];   // padded stride: 80B rows
  __shared__ __align__(16) bf16 Bs[128][40];

  const int tid  = threadIdx.x;   // 0..127
  const int lane = tid & 31;
  const int wave = tid >> 5;      // 0..3
  const int wm   = wave >> 1;     // 0..1 (M)
  const int wn   = wave & 1;      // 0..1 (N)
  const int half = lane >> 4;
  const int nl   = lane & 15;

  const int blockM = blockIdx.x * 128;
  const int blockN = blockIdx.y * 128;

  // LDS fill: lane covers (row = r4*32 + tid/4, 16B chunk = tid%4) so
  // consecutive lanes fetch consecutive 16B chunks of the same row slab.
  const int lrow = tid >> 2;        // 0..31
  const int lce  = (tid & 3) * 8;   // 0,8,16,24 (bf16 elements)

  v8f acc[4][4];
#pragma unroll
  for (int i = 0; i < 4; ++i)
#pragma unroll
    for (int j = 0; j < 4; ++j)
      acc[i][j] = (v8f){0.f,0.f,0.f,0.f,0.f,0.f,0.f,0.f};

  const int klo = half ? 8 : 0;     // per-lane k offset inside 32-wide slab

  for (int k0 = 0; k0 < K; k0 += 32) {
    // ---- stage 128x32 A and B slabs into LDS ----
#pragma unroll
    for (int r4 = 0; r4 < 4; ++r4) {
      const int row = r4 * 32 + lrow;
      g2l_b128(&As[row][lce], A  + (size_t)(blockM + row) * K + k0 + lce);
      g2l_b128(&Bs[row][lce], Bw + (size_t)(blockN + row) * K + k0 + lce);
    }
    g2l_fence();
    __syncthreads();

    // ---- fragments: A lane = M row (k-contiguous); B mirrors with N ----
    v16bf af[4];
#pragma unroll
    for (int i = 0; i < 4; ++i) {
      const int r = wm * 64 + i * 16 + nl;
      af[i] = combine16(*(const v8bf*)&As[r][klo],
                        *(const v8bf*)&As[r][klo + 16]);
    }
#pragma unroll
    for (int j = 0; j < 4; ++j) {
      const int c = wn * 64 + j * 16 + nl;
      v16bf bfr = combine16(*(const v8bf*)&Bs[c][klo],
                            *(const v8bf*)&Bs[c][klo + 16]);
#pragma unroll
      for (int i = 0; i < 4; ++i)
        acc[i][j] = wmma_bf16(af[i], bfr, acc[i][j]);
    }
    __syncthreads();
  }

  // ---- epilogue: C layout = VGPR v -> M = v + 8*half, lane -> N = nl ----
#pragma unroll
  for (int i = 0; i < 4; ++i) {
#pragma unroll
    for (int j = 0; j < 4; ++j) {
#pragma unroll
      for (int v = 0; v < 8; ++v) {
        const int gm = blockM + wm * 64 + i * 16 + v + half * 8;
        const int gn = blockN + wn * 64 + j * 16 + nl;
        const float val = (acc[i][j][v] + bias[gn]) * alpha;
        if (MODE == 0) {
          ((float*)outv)[(size_t)gm * N + gn] = val;
        } else {
          const int b = gm >> 11, s = gm & (SDIM - 1);
          const int h = gn >> 6,  d = gn & (HDDIM - 1);
          ((bf16*)outv)[((size_t)((b * HDIM + h) * SDIM + s)) * HDDIM + d] =
              (bf16)val;
        }
      }
    }
  }
}

// ---------------------------------------------------------------------------
// Flash attention per (b,h). Block = 256 threads = 8 waves; each wave owns 16
// query rows. Keys streamed 32 at a time: K async-staged row-major, V staged
// transposed so the P*V B-operand is k-contiguous. Online softmax: row max by
// shfl_xor inside 16-lane halves; row SUM computed as one extra WMMA against
// an all-ones B operand (rowsum = P x 1), which lands broadcast across lanes
// in exactly the C-layout register mapping. Q arrives pre-scaled by 1/sqrt(HD).
// ---------------------------------------------------------------------------
__global__ __launch_bounds__(256)
void flash_attn_kernel(const bf16* __restrict__ q, const bf16* __restrict__ k,
                       const bf16* __restrict__ v, const float* __restrict__ sumv,
                       bf16* __restrict__ ctx) {
  __shared__ __align__(16) bf16 Kt[32][72];      // 32 keys x 64 hd (padded)
  __shared__ __align__(16) bf16 Vt[64][40];      // transposed: d x key
  __shared__ __align__(16) bf16 Pl[8][16][40];   // per-wave P bounce buffer

  const int tid  = threadIdx.x;
  const int lane = tid & 31;
  const int w    = tid >> 5;
  const int half = lane >> 4;
  const int nl   = lane & 15;

  const int bh = blockIdx.x >> 4;               // (b*H + h), S/128 = 16 q-tiles
  const int qt = blockIdx.x & 15;
  const int q0 = qt * 128 + w * 16;

  const bf16* qp = q + (size_t)bh * SDIM * HDDIM;
  const bf16* kp = k + (size_t)bh * SDIM * HDDIM;
  const bf16* vp = v + (size_t)bh * SDIM * HDDIM;

  const int klo = half ? 8 : 0;

  // Q fragments for this wave's 16 rows (2 chunks of HD=64), kept in regs.
  v16bf qfrag[2];
  {
    const size_t r = (size_t)(q0 + nl) * HDDIM;
#pragma unroll
    for (int c = 0; c < 2; ++c) {
      const int kq = c * 32 + klo;
      qfrag[c] = combine16(*(const v8bf*)(qp + r + kq),
                           *(const v8bf*)(qp + r + kq + 16));
    }
  }

  // All-ones B operand for the rowsum WMMA (layout-independent).
  v16bf onesf;
#pragma unroll
  for (int i = 0; i < 16; ++i) onesf[i] = (bf16)1.0f;

  float rowmax[8], rowsum[8], cf[8];
#pragma unroll
  for (int i = 0; i < 8; ++i) { rowmax[i] = -1e30f; rowsum[i] = 0.f; }
  v8f o[4];
#pragma unroll
  for (int j = 0; j < 4; ++j) o[j] = (v8f){0.f,0.f,0.f,0.f,0.f,0.f,0.f,0.f};

  const int ldr = tid >> 3;        // 0..31  key row
  const int ldc = (tid & 7) * 8;   // 0..56  hd col (8 bf16 chunk)

  for (int kb = 0; kb < SDIM; kb += 32) {
    // ---- stage K (async, row-major) and V (register transpose) ----
    {
      g2l_b128(&Kt[ldr][ldc], kp + (size_t)(kb + ldr) * HDDIM + ldc);
      v8bf vr = *(const v8bf*)(vp + (size_t)(kb + ldr) * HDDIM + ldc);
#pragma unroll
      for (int e = 0; e < 8; ++e) Vt[ldc + e][ldr] = vr[e];
    }
    g2l_fence();
    __syncthreads();

    // ---- scores: S[16q x 32k] = Q * K^T (already includes 1/sqrt(HD)) ----
    v8f s0 = (v8f){0.f,0.f,0.f,0.f,0.f,0.f,0.f,0.f};
    v8f s1 = s0;
#pragma unroll
    for (int c = 0; c < 2; ++c) {
      const int kq = c * 32 + klo;
      v16bf kf0 = combine16(*(const v8bf*)&Kt[nl][kq],
                            *(const v8bf*)&Kt[nl][kq + 16]);
      v16bf kf1 = combine16(*(const v8bf*)&Kt[16 + nl][kq],
                            *(const v8bf*)&Kt[16 + nl][kq + 16]);
      s0 = wmma_bf16(qfrag[c], kf0, s0);
      s1 = wmma_bf16(qfrag[c], kf1, s1);
    }

    // ---- online softmax: max via shfl_xor, exp, stash P in LDS ----
#pragma unroll
    for (int r = 0; r < 8; ++r) {
      const float a0 = s0[r];
      const float a1 = s1[r];
      float mx = fmaxf(a0, a1);
      mx = fmaxf(mx, __shfl_xor(mx, 1, 32));
      mx = fmaxf(mx, __shfl_xor(mx, 2, 32));
      mx = fmaxf(mx, __shfl_xor(mx, 4, 32));
      mx = fmaxf(mx, __shfl_xor(mx, 8, 32));
      const float mnew = fmaxf(rowmax[r], mx);
      cf[r] = __expf(rowmax[r] - mnew);
      rowmax[r] = mnew;
      const float p0 = __expf(a0 - mnew);
      const float p1 = __expf(a1 - mnew);
#pragma unroll
      for (int j = 0; j < 4; ++j) o[j][r] *= cf[r];
      const int m = r + half * 8;           // C-layout row
      Pl[w][m][nl]      = (bf16)p0;
      Pl[w][m][16 + nl] = (bf16)p1;
    }

    // ---- reload P in A-layout (wave-local LDS bounce) ----
    v16bf pf = combine16(*(const v8bf*)&Pl[w][nl][klo],
                         *(const v8bf*)&Pl[w][nl][klo + 16]);

    // ---- rowsum = P x ones via WMMA (replaces 32 ds_bpermute per tile) ----
    v8f rs8 = wmma_bf16(pf, onesf,
                        (v8f){0.f,0.f,0.f,0.f,0.f,0.f,0.f,0.f});
#pragma unroll
    for (int r = 0; r < 8; ++r) rowsum[r] = rowsum[r] * cf[r] + rs8[r];

    // ---- O += P * V ----
#pragma unroll
    for (int j = 0; j < 4; ++j) {
      const int d = j * 16 + nl;
      v16bf vf = combine16(*(const v8bf*)&Vt[d][klo],
                           *(const v8bf*)&Vt[d][klo + 16]);
      o[j] = wmma_bf16(pf, vf, o[j]);
    }
    __syncthreads();
  }

  // ---- epilogue: out = O / L + 1e-8 * sumV, stored into ctx[B,S,E] ----
  const int b = bh >> 4, h = bh & (HDIM - 1);
#pragma unroll
  for (int r = 0; r < 8; ++r) {
    const int srow = q0 + r + half * 8;
    const float inv = 1.0f / rowsum[r];
#pragma unroll
    for (int j = 0; j < 4; ++j) {
      const int d = j * 16 + nl;
      const float val = o[j][r] * inv + 1e-8f * sumv[bh * HDDIM + d];
      ctx[((size_t)(b * SDIM + srow)) * EDIM + h * HDDIM + d] = (bf16)val;
    }
  }
}

// ---------------------------------------------------------------------------
// Host-side orchestration
// ---------------------------------------------------------------------------
extern "C" void kernel_launch(void* const* d_in, const int* in_sizes, int n_in,
                              void* d_out, int out_size, void* d_ws, size_t ws_size,
                              hipStream_t stream) {
  (void)in_sizes; (void)n_in; (void)out_size; (void)ws_size;
  const float* x  = (const float*)d_in[0];
  const float* wq = (const float*)d_in[1];
  const float* bq = (const float*)d_in[2];
  const float* wk = (const float*)d_in[3];
  const float* bk = (const float*)d_in[4];
  const float* wv = (const float*)d_in[5];
  const float* bv = (const float*)d_in[6];
  const float* wo = (const float*)d_in[7];
  const float* bo = (const float*)d_in[8];
  float* out = (float*)d_out;

  // Workspace carve-up (all 16B aligned): ~48 MiB total
  char* ws = (char*)d_ws;
  size_t off = 0;
  bf16* xb   = (bf16*)(ws + off); off += (size_t)MROWS * EDIM * 2;  // 8 MiB
  bf16* wqb  = (bf16*)(ws + off); off += (size_t)EDIM * EDIM * 2;   // 2 MiB
  bf16* wkb  = (bf16*)(ws + off); off += (size_t)EDIM * EDIM * 2;
  bf16* wvb  = (bf16*)(ws + off); off += (size_t)EDIM * EDIM * 2;
  bf16* wob  = (bf16*)(ws + off); off += (size_t)EDIM * EDIM * 2;
  bf16* qb   = (bf16*)(ws + off); off += (size_t)MROWS * EDIM * 2;  // [B,H,S,HD]
  bf16* kb   = (bf16*)(ws + off); off += (size_t)MROWS * EDIM * 2;
  bf16* vb   = (bf16*)(ws + off); off += (size_t)MROWS * EDIM * 2;
  bf16* ctx  = (bf16*)(ws + off); off += (size_t)MROWS * EDIM * 2;  // [B,S,E]
  float* sv  = (float*)(ws + off); off += (size_t)BDIM * HDIM * HDDIM * 4;

  // 1) downcast activations + weights to bf16
  const int nx = MROWS * EDIM, nw = EDIM * EDIM;
  cvt_bf16_kernel<<<2048, 256, 0, stream>>>(x,  xb,  nx);
  cvt_bf16_kernel<<<1024, 256, 0, stream>>>(wq, wqb, nw);
  cvt_bf16_kernel<<<1024, 256, 0, stream>>>(wk, wkb, nw);
  cvt_bf16_kernel<<<1024, 256, 0, stream>>>(wv, wvb, nw);
  cvt_bf16_kernel<<<1024, 256, 0, stream>>>(wo, wob, nw);

  // 2) QKV projections (WMMA GEMM, scatter into [B,H,S,HD]).
  //    Q gets alpha = 1/sqrt(HD) so the flash kernel does no score scaling.
  dim3 ggrid(MROWS / 128, EDIM / 128);  // 32 x 8
  gemm_bf16_kernel<1><<<ggrid, 128, 0, stream>>>(xb, wqb, bq, 0.125f, qb,
                                                 MROWS, EDIM, EDIM);
  gemm_bf16_kernel<1><<<ggrid, 128, 0, stream>>>(xb, wkb, bk, 1.0f, kb,
                                                 MROWS, EDIM, EDIM);
  gemm_bf16_kernel<1><<<ggrid, 128, 0, stream>>>(xb, wvb, bv, 1.0f, vb,
                                                 MROWS, EDIM, EDIM);

  // 3) sum_k V (for the reference's softmax + 1e-8 term)
  sumv_kernel<<<BDIM * HDIM, HDDIM, 0, stream>>>(vb, sv);

  // 4) flash attention -> ctx [B,S,E] bf16
  flash_attn_kernel<<<BDIM * HDIM * (SDIM / 128), 256, 0, stream>>>(qb, kb, vb,
                                                                    sv, ctx);

  // 5) output projection -> f32 d_out
  gemm_bf16_kernel<0><<<ggrid, 128, 0, stream>>>(ctx, wob, bo, 1.0f, out,
                                                 MROWS, EDIM, EDIM);
}
